// Eagle3DraftModel_82652350644927
// MI455X (gfx1250) — matrix operand order repaired and checked by reading, not compiled, not run on hardware
//
#include <hip/hip_runtime.h>
#include <cstdint>

// ---------------- problem constants (from reference) ----------------
#define BB   2
#define SS   1024
#define HH   2048
#define NHH  32
#define NKVV 8
#define DD   64
#define II   8192
#define LCKK 3

typedef __bf16 bf16_t;
typedef __bf16 v16bf __attribute__((ext_vector_type(16)));
typedef float  v8f   __attribute__((ext_vector_type(8)));

// ---------------- helpers ----------------
__device__ inline unsigned short bfbits(float f) {
  unsigned u = __float_as_uint(f);
  u = (u + 0x7FFFu + ((u >> 16) & 1u)) >> 16;   // round-to-nearest-even
  return (unsigned short)u;
}
__device__ inline bf16_t tobf(float f) {
  unsigned short s = bfbits(f);
  return __builtin_bit_cast(bf16_t, s);
}
__device__ inline unsigned bfpack2(float lo, float hi) {
  return (unsigned)bfbits(lo) | ((unsigned)bfbits(hi) << 16);
}

union Frag16 {             // one WMMA 16-bit operand: 8 VGPRs / lane
  v16bf v;
  uint4 q[2];
  unsigned u[8];
};

// reduce across the 16 lanes of a half-wave (wave32: masks < 16 stay in-half)
__device__ inline float redmax16(float v) {
#pragma unroll
  for (int m = 8; m >= 1; m >>= 1) v = fmaxf(v, __shfl_xor(v, m, 32));
  return v;
}
__device__ inline float redsum16(float v) {
#pragma unroll
  for (int m = 8; m >= 1; m >>= 1) v += __shfl_xor(v, m, 32);
  return v;
}

// =====================================================================
// Vectorized f32 -> bf16 conversion (4 elements / thread)
// =====================================================================
__global__ __launch_bounds__(256) void cvt_bf16x4_kernel(
    const float4* __restrict__ in, uint2* __restrict__ out, int n4)
{
  int i = blockIdx.x * 256 + threadIdx.x;
  if (i < n4) {
    float4 f = in[i];
    uint2 r;
    r.x = bfpack2(f.x, f.y);
    r.y = bfpack2(f.z, f.w);
    out[i] = r;
  }
}

// =====================================================================
// GEMM:  C[M,N] (f32) = A[M,K] (bf16 row-major) * W[N,K]^T (bf16 row-major)
// per the einsum 'bsi,oi->bso'.
// Block = 4 waves (M-split); the shared 64x32 W tile is staged into LDS
// once per block with async-to-LDS (ASYNCcnt), double-buffered so the
// copy of step k+1 overlaps the WMMAs of step k.
// =====================================================================
__global__ __launch_bounds__(128) void gemm_bf16_wmma(
    const bf16_t* __restrict__ A, const bf16_t* __restrict__ W,
    float* __restrict__ C, int M, int N, int K)
{
  __shared__ __align__(16) bf16_t sW[2][64 * 32];   // 2 x 4 KB
  const int tid  = threadIdx.x;
  const int lane = tid & 31;
  const int wave = tid >> 5;
  const int hf = lane >> 4;           // lane half
  const int li = lane & 15;
  const int mBase = blockIdx.x * 64 + wave * 16;
  const int nBase = blockIdx.y * 64;

  v8f acc[4];
#pragma unroll
  for (int o = 0; o < 4; ++o)
#pragma unroll
    for (int r = 0; r < 8; ++r) acc[o][r] = 0.f;

  // A fragment: lane holds row (mBase+li); K range hf*16 .. hf*16+15
  const bf16_t* arow = A + (size_t)(mBase + li) * K + hf * 16;

  // staging assignment: each of the 128 threads copies 32 bytes of the tile
  const int srow  = tid >> 1;   // 0..63 : W row within tile
  const int shalf = tid & 1;    // K elements 0-15 / 16-31
  const bf16_t* wsrc = W + (size_t)(nBase + srow) * K + shalf * 16;
  const unsigned ldsTileOff = (unsigned)(srow * 64 + shalf * 32);   // bytes
  const unsigned ldsBase0 = (unsigned)(uintptr_t)(&sW[0][0]) + ldsTileOff;
  const unsigned ldsBase1 = (unsigned)(uintptr_t)(&sW[1][0]) + ldsTileOff;

  // prologue: async-stage kk=0 into buffer 0 (32B per lane = two b128)
  asm volatile("global_load_async_to_lds_b128 %0, %1, off\n\t"
               "global_load_async_to_lds_b128 %0, %1, off offset:16"
               :: "v"(ldsBase0), "v"(wsrc) : "memory");

  int p = 0;
  for (int kk = 0; kk < K; kk += 32, p ^= 1) {
    asm volatile("s_wait_asynccnt 0x0" ::: "memory");
    __syncthreads();   // buffer p complete; prev reads finished before overwrite
    if (kk + 32 < K) {
      unsigned l = p ? ldsBase0 : ldsBase1;
      const bf16_t* g = wsrc + kk + 32;
      asm volatile("global_load_async_to_lds_b128 %0, %1, off\n\t"
                   "global_load_async_to_lds_b128 %0, %1, off offset:16"
                   :: "v"(l), "v"(g) : "memory");
    }
    Frag16 a;
    a.q[0] = *(const uint4*)(arow + kk);
    a.q[1] = *(const uint4*)(arow + kk + 8);
    const bf16_t* wt = &sW[p][0];
#pragma unroll
    for (int o = 0; o < 4; ++o) {
      // B fragment: lane holds column n = li -> tile row (o*16+li),
      // K range hf*16 .. hf*16+15 (32B contiguous in LDS)
      const bf16_t* wr = wt + (o * 16 + li) * 32 + hf * 16;
      Frag16 bfr;
      bfr.q[0] = *(const uint4*)(wr);
      bfr.q[1] = *(const uint4*)(wr + 8);
      acc[o] = __builtin_amdgcn_wmma_f32_16x16x32_bf16(
          false, a.v, false, bfr.v, (short)0, acc[o], false, false);
    }
  }

#pragma unroll
  for (int o = 0; o < 4; ++o)
#pragma unroll
    for (int r = 0; r < 8; ++r) {
      int row = mBase + r + 8 * hf;     // C/D layout: VGPR r -> rows r / r+8
      int col = nBase + o * 16 + li;
      C[(size_t)row * N + col] = acc[o][r];
    }
}

// =====================================================================
// Flash attention with diagonal extras (one wave = one 16-row q tile).
// s0: causal attention vs cache_k[0]; extras: cache_k[1],cache_k[2],fresh k
// =====================================================================
__global__ __launch_bounds__(32) void flash_attn_kernel(
    const bf16_t* __restrict__ Qb,     // [B,NH,S,D] bf16 (post-RoPE)
    const bf16_t* __restrict__ K0,     // [B,NH,S,D] bf16 (cache_k[0])
    const bf16_t* __restrict__ V0T,    // [B,NH,D,S] bf16 (cache_v[0]^T)
    const float*  __restrict__ cacheK, // [LCK,B,NH,S,D] f32
    const float*  __restrict__ cacheV,
    const bf16_t* __restrict__ Kf,     // [B,NKV,S,D] fresh K (post-RoPE)
    const bf16_t* __restrict__ Vf,     // [B,NKV,S,D] fresh V
    bf16_t* __restrict__ Out)          // [B*S, NH*D]
{
  __shared__ __align__(16) bf16_t sP[16 * 32];
  const int lane = threadIdx.x;
  const int hf = lane >> 4;
  const int li = lane & 15;
  const int qt = blockIdx.x, h = blockIdx.y, b = blockIdx.z;
  const int qbase = qt * 16;
  const size_t bh = (size_t)b * NHH + h;
  const float scale = 0.125f;   // 1/sqrt(64)

  // Q fragments for K-dim chunks 0-31 and 32-63
  const bf16_t* qtile = Qb + (bh * SS + qbase) * DD;
  Frag16 a0, a1;
  {
    const bf16_t* ar = qtile + (size_t)li * DD + hf * 16;
    a0.q[0] = *(const uint4*)(ar);      a0.q[1] = *(const uint4*)(ar + 8);
    a1.q[0] = *(const uint4*)(ar + 32); a1.q[1] = *(const uint4*)(ar + 40);
  }

  // ---- diagonal extras: per-row dot products (2 lanes/row, 32 dims each) ----
  const int rowg = qbase + li;
  float qv[32];
  {
    const bf16_t* qr = qtile + (size_t)li * DD + hf * 32;
#pragma unroll
    for (int t = 0; t < 32; ++t) qv[t] = (float)qr[t];
  }
  float sx0, sx1, sx2;
  {
    const float* k1 = cacheK + (((size_t)1 * BB + b) * NHH + h) * SS * DD + (size_t)rowg * DD + hf * 32;
    const float* k2 = cacheK + (((size_t)2 * BB + b) * NHH + h) * SS * DD + (size_t)rowg * DD + hf * 32;
    const bf16_t* k3 = Kf + (((size_t)b * NKVV + (h >> 2)) * SS + rowg) * DD + hf * 32;
    float d0 = 0.f, d1 = 0.f, d2 = 0.f;
#pragma unroll
    for (int t = 0; t < 32; ++t) {
      d0 += qv[t] * k1[t]; d1 += qv[t] * k2[t]; d2 += qv[t] * (float)k3[t];
    }
    d0 += __shfl_xor(d0, 16, 32);
    d1 += __shfl_xor(d1, 16, 32);
    d2 += __shfl_xor(d2, 16, 32);
    sx0 = d0 * scale; sx1 = d1 * scale; sx2 = d2 * scale;
  }
  float m0 = fmaxf(sx0, fmaxf(sx1, sx2));
  float p0 = __expf(sx0 - m0), p1 = __expf(sx1 - m0), p2 = __expf(sx2 - m0);
  float l0 = p0 + p1 + p2;

  float m_st[8], l_st[8];
  v8f acc[4];
#pragma unroll
  for (int o = 0; o < 4; ++o)
#pragma unroll
    for (int r = 0; r < 8; ++r) acc[o][r] = 0.f;

#pragma unroll
  for (int r = 0; r < 8; ++r) {
    int src = r + 8 * hf;                 // row owned by this (slot, half)
    m_st[r] = __shfl(m0, src, 32);
    l_st[r] = __shfl(l0, src, 32);
    float q0 = __shfl(p0, src, 32);
    float q1 = __shfl(p1, src, 32);
    float q2 = __shfl(p2, src, 32);
    int rr = qbase + r + 8 * hf;
    const float* v1 = cacheV + (((size_t)1 * BB + b) * NHH + h) * SS * DD + (size_t)rr * DD;
    const float* v2 = cacheV + (((size_t)2 * BB + b) * NHH + h) * SS * DD + (size_t)rr * DD;
    const bf16_t* v3 = Vf + (((size_t)b * NKVV + (h >> 2)) * SS + rr) * DD;
#pragma unroll
    for (int o = 0; o < 4; ++o) {
      int d = o * 16 + li;
      acc[o][r] = q0 * v1[d] + q1 * v2[d] + q2 * (float)v3[d];
    }
  }

  const bf16_t* ktile = K0 + bh * SS * DD;
  const bf16_t* vtt = V0T + bh * (size_t)DD * SS;

  // ---- causal flash loop over cache_k[0]/cache_v[0], 32 keys/iter ----
  for (int kb = 0; kb < qbase + 16; kb += 32) {
    Frag16 b00, b01, b10, b11;
    {
      const bf16_t* kr = ktile + (size_t)(kb + li) * DD + hf * 16;
      b00.q[0] = *(const uint4*)(kr);      b00.q[1] = *(const uint4*)(kr + 8);
      b01.q[0] = *(const uint4*)(kr + 32); b01.q[1] = *(const uint4*)(kr + 40);
      const bf16_t* kr2 = kr + 16 * DD;
      b10.q[0] = *(const uint4*)(kr2);      b10.q[1] = *(const uint4*)(kr2 + 8);
      b11.q[0] = *(const uint4*)(kr2 + 32); b11.q[1] = *(const uint4*)(kr2 + 40);
    }
    v8f z = {0.f, 0.f, 0.f, 0.f, 0.f, 0.f, 0.f, 0.f};
    v8f slo = __builtin_amdgcn_wmma_f32_16x16x32_bf16(false, a0.v, false, b00.v, (short)0, z,   false, false);
    slo     = __builtin_amdgcn_wmma_f32_16x16x32_bf16(false, a1.v, false, b01.v, (short)0, slo, false, false);
    v8f shi = __builtin_amdgcn_wmma_f32_16x16x32_bf16(false, a0.v, false, b10.v, (short)0, z,   false, false);
    shi     = __builtin_amdgcn_wmma_f32_16x16x32_bf16(false, a1.v, false, b11.v, (short)0, shi, false, false);

    __syncthreads();   // protect sP vs previous iteration's reads

    const int keyL = kb + li;
    const int keyH = kb + 16 + li;
#pragma unroll
    for (int r = 0; r < 8; ++r) {
      const int row = qbase + r + 8 * hf;
      float xl = (keyL <= row) ? slo[r] * scale : -3.0e38f;
      float xh = (keyH <= row) ? shi[r] * scale : -3.0e38f;
      float mx = redmax16(fmaxf(xl, xh));
      float mnew = fmaxf(m_st[r], mx);
      float sc_old = __expf(m_st[r] - mnew);  // m_st finite from extras init
      float pl = __expf(xl - mnew);
      float ph = __expf(xh - mnew);
      float rs = redsum16(pl + ph);
      l_st[r] = l_st[r] * sc_old + rs;
      m_st[r] = mnew;
#pragma unroll
      for (int o = 0; o < 4; ++o) acc[o][r] *= sc_old;
      sP[(r + 8 * hf) * 32 + li]      = tobf(pl);
      sP[(r + 8 * hf) * 32 + 16 + li] = tobf(ph);
    }
    __syncthreads();

    // re-shape P into A-fragment layout, then P @ V (4 output dim-tiles)
    Frag16 ap;
    ap.q[0] = *(const uint4*)(&sP[li * 32 + hf * 16]);
    ap.q[1] = *(const uint4*)(&sP[li * 32 + hf * 16 + 8]);
#pragma unroll
    for (int o = 0; o < 4; ++o) {
      const bf16_t* vr = vtt + (size_t)(o * 16 + li) * SS + kb + hf * 16;
      Frag16 bv;
      bv.q[0] = *(const uint4*)(vr);
      bv.q[1] = *(const uint4*)(vr + 8);
      acc[o] = __builtin_amdgcn_wmma_f32_16x16x32_bf16(
          false, ap.v, false, bv.v, (short)0, acc[o], false, false);
    }
  }

  // epilogue: normalize and store bf16 [B*S, NH*D]
#pragma unroll
  for (int r = 0; r < 8; ++r) {
    float il = 1.0f / l_st[r];
    int rr = qbase + r + 8 * hf;
    bf16_t* orow = Out + ((size_t)b * SS + rr) * (NHH * DD) + h * DD;
#pragma unroll
    for (int o = 0; o < 4; ++o)
      orow[o * 16 + li] = tobf(acc[o][r] * il);
  }
}

// =====================================================================
// RMS norm kernels
// =====================================================================
__global__ __launch_bounds__(256) void rmsnorm_concat_kernel(
    const float* __restrict__ emb, const float* __restrict__ hid,
    const float* __restrict__ w_e, const float* __restrict__ w_h,
    bf16_t* __restrict__ X)   // [B*S, 2H] = [rms(e)*we , rms(h)*wh]
{
  __shared__ float red[256];
  const int row = blockIdx.x;
  const int t = threadIdx.x;
  const float* e = emb + (size_t)row * HH;
  const float* h = hid + (size_t)row * HH;
  float ve[8], vh[8];
  float se = 0.f, sh = 0.f;
#pragma unroll
  for (int j = 0; j < 8; ++j) {
    int c = t + j * 256;
    ve[j] = e[c]; vh[j] = h[c];
    se += ve[j] * ve[j]; sh += vh[j] * vh[j];
  }
  red[t] = se; __syncthreads();
  for (int s = 128; s > 0; s >>= 1) { if (t < s) red[t] += red[t + s]; __syncthreads(); }
  float seTot = red[0]; __syncthreads();
  red[t] = sh; __syncthreads();
  for (int s = 128; s > 0; s >>= 1) { if (t < s) red[t] += red[t + s]; __syncthreads(); }
  float shTot = red[0];
  float re = rsqrtf(seTot / (float)HH + 1e-6f);
  float rh = rsqrtf(shTot / (float)HH + 1e-6f);
  bf16_t* xr = X + (size_t)row * (2 * HH);
#pragma unroll
  for (int j = 0; j < 8; ++j) {
    int c = t + j * 256;
    xr[c]      = tobf(ve[j] * re * w_e[c]);
    xr[HH + c] = tobf(vh[j] * rh * w_h[c]);
  }
}

__global__ __launch_bounds__(256) void rmsnorm_kernel(
    const float* __restrict__ in, const float* __restrict__ w,
    bf16_t* __restrict__ out)
{
  __shared__ float red[256];
  const int row = blockIdx.x;
  const int t = threadIdx.x;
  const float* x = in + (size_t)row * HH;
  float v[8]; float s = 0.f;
#pragma unroll
  for (int j = 0; j < 8; ++j) { int c = t + j * 256; v[j] = x[c]; s += v[j] * v[j]; }
  red[t] = s; __syncthreads();
  for (int k = 128; k > 0; k >>= 1) { if (t < k) red[t] += red[t + k]; __syncthreads(); }
  float r = rsqrtf(red[0] / (float)HH + 1e-6f);
#pragma unroll
  for (int j = 0; j < 8; ++j) { int c = t + j * 256; out[(size_t)row * HH + c] = tobf(v[j] * r * w[c]); }
}

// =====================================================================
// RoPE + layout change: [B,S,NH,D]f32 -> [B,NH,S,D]bf16 (and K similarly)
// =====================================================================
__global__ __launch_bounds__(256) void rope_kernel(
    const float* __restrict__ qf, const float* __restrict__ kf,
    bf16_t* __restrict__ qb, bf16_t* __restrict__ kb)
{
  int idx = blockIdx.x * 256 + threadIdx.x;
  const int TOT = BB * SS * (NHH + NKVV) * 32;
  if (idx >= TOT) return;
  int i = idx & 31;
  int hr = idx >> 5;
  int head = hr % (NHH + NKVV);
  int row = hr / (NHH + NKVV);      // b*S + s
  int s = row % SS, b = row / SS;
  float inv = __powf(10000.f, -(float)i * (1.0f / 32.0f));
  float ang = (float)(s + LCKK) * inv;
  float sn, cs; __sincosf(ang, &sn, &cs);
  const float* src; bf16_t* dst;
  if (head < NHH) {
    src = qf + (size_t)row * (NHH * DD) + head * DD;
    dst = qb + (((size_t)b * NHH + head) * SS + s) * DD;
  } else {
    int kv = head - NHH;
    src = kf + (size_t)row * (NKVV * DD) + kv * DD;
    dst = kb + (((size_t)b * NKVV + kv) * SS + s) * DD;
  }
  float x0 = src[i], x1 = src[i + 32];
  dst[i]      = tobf(x0 * cs - x1 * sn);
  dst[i + 32] = tobf(x1 * cs + x0 * sn);
}

// fresh V: [B,S,NKV,D]f32 -> [B,NKV,S,D]bf16
__global__ __launch_bounds__(256) void vtrans_kernel(
    const float* __restrict__ vf, bf16_t* __restrict__ vb)
{
  int idx = blockIdx.x * 256 + threadIdx.x;
  const int TOT = BB * SS * NKVV * DD;
  if (idx >= TOT) return;
  int d = idx & 63;
  int kv = (idx >> 6) & 7;
  int row = idx >> 9;               // b*S + s
  int s = row & (SS - 1), b = row >> 10;
  vb[(((size_t)b * NKVV + kv) * SS + s) * DD + d] = tobf(vf[idx]);
}

// cache_v[0]: [B,NH,S,D]f32 -> [B,NH,D,S]bf16  (transposed for PV B-frags)
__global__ __launch_bounds__(256) void v0t_kernel(
    const float* __restrict__ v0, bf16_t* __restrict__ vt)
{
  int idx = blockIdx.x * 256 + threadIdx.x;
  const int TOT = BB * NHH * SS * DD;
  if (idx >= TOT) return;
  int d = idx & 63;
  int s = (idx >> 6) & (SS - 1);
  int bhh = idx >> 16;
  vt[((size_t)bhh * DD + d) * SS + s] = tobf(v0[idx]);
}

__global__ __launch_bounds__(256) void add_kernel(
    const float* __restrict__ a, const float* __restrict__ b,
    float* __restrict__ c, int n)
{
  int i = blockIdx.x * 256 + threadIdx.x;
  if (i < n) c[i] = a[i] + b[i];
}

__global__ __launch_bounds__(256) void silu_mul_kernel(
    const float* __restrict__ g, const float* __restrict__ u,
    bf16_t* __restrict__ act, int n)
{
  int i = blockIdx.x * 256 + threadIdx.x;
  if (i < n) {
    float x = g[i];
    float sg = 1.0f / (1.0f + __expf(-x));
    act[i] = tobf(x * sg * u[i]);
  }
}

// =====================================================================
// Host orchestration
// =====================================================================
extern "C" void kernel_launch(void* const* d_in, const int* in_sizes, int n_in,
                              void* d_out, int out_size, void* d_ws, size_t ws_size,
                              hipStream_t stream)
{
  (void)in_sizes; (void)n_in; (void)out_size; (void)ws_size;
  const float* input_emb = (const float*)d_in[0];
  const float* hidden    = (const float*)d_in[1];
  const float* cache_k   = (const float*)d_in[2];
  const float* cache_v   = (const float*)d_in[3];
  // d_in[4] attention_mask (exact causal, applied analytically)
  // d_in[5] position_ids   (arange(S), applied analytically)
  const float* wq = (const float*)d_in[6];
  const float* wk = (const float*)d_in[7];
  const float* wv = (const float*)d_in[8];
  const float* wo = (const float*)d_in[9];
  const float* wg = (const float*)d_in[10];
  const float* wu = (const float*)d_in[11];
  const float* wd = (const float*)d_in[12];
  const float* w_hid_norm = (const float*)d_in[13];
  const float* w_in_ln    = (const float*)d_in[14];
  const float* w_post_ln  = (const float*)d_in[15];
  float* out = (float*)d_out;

  char* ws = (char*)d_ws;
  size_t off = 0;
  auto alloc = [&](size_t bytes) -> char* {
    char* p = ws + off;
    off = (off + bytes + 255) & ~(size_t)255;
    return p;
  };
  const int M = BB * SS;   // 2048 token rows
  bf16_t* xbf   = (bf16_t*)alloc((size_t)M * 4096 * 2);  // concat norm out
  bf16_t* qbf   = (bf16_t*)alloc((size_t)M * 2048 * 2);  // [B,NH,S,D]
  bf16_t* kfbf  = (bf16_t*)alloc((size_t)M * 512 * 2);   // [B,NKV,S,D]
  bf16_t* vfbf  = (bf16_t*)alloc((size_t)M * 512 * 2);
  bf16_t* k0bf  = (bf16_t*)alloc((size_t)M * 2048 * 2);  // cache_k[0] bf16
  bf16_t* v0t   = (bf16_t*)alloc((size_t)M * 2048 * 2);  // cache_v[0]^T bf16
  bf16_t* attnb = (bf16_t*)alloc((size_t)M * 2048 * 2);  // attention out bf16
  float*  h2    = (float*) alloc((size_t)M * 2048 * 4);  // residual + attn
  bf16_t* hnbf  = (bf16_t*)alloc((size_t)M * 2048 * 2);  // post-norm bf16
  bf16_t* wbuf  = (bf16_t*)alloc((size_t)II * HH * 2);   // bf16 weight scratch (reused)
  float*  t0    = (float*) alloc((size_t)M * 8192 * 4);  // big transient 0
  float*  t1    = (float*) alloc((size_t)M * 8192 * 4);  // big transient 1
  bf16_t* act   = (bf16_t*)alloc((size_t)M * 8192 * 2);  // silu(g)*u bf16

  float* qf32 = t0;                          // [M, 2048] pre-RoPE
  float* kf32 = t0 + (size_t)M * 2048;       // [M, 512]
  float* vf32 = kf32 + (size_t)M * 512;      // [M, 512]

  auto cvtW = [&](const float* w, size_t elems) {
    int n4 = (int)(elems / 4);
    cvt_bf16x4_kernel<<<(n4 + 255) / 256, 256, 0, stream>>>(
        (const float4*)w, (uint2*)wbuf, n4);
  };

  // 1) norms + concat -> x (bf16)
  rmsnorm_concat_kernel<<<M, 256, 0, stream>>>(input_emb, hidden, w_in_ln, w_hid_norm, xbf);

  // 2) QKV projections (WMMA, bf16 weights staged once per matrix)
  cvtW(wq, (size_t)2048 * 4096);
  gemm_bf16_wmma<<<dim3(M / 64, 2048 / 64), 128, 0, stream>>>(xbf, wbuf, qf32, M, 2048, 4096);
  cvtW(wk, (size_t)512 * 4096);
  gemm_bf16_wmma<<<dim3(M / 64,  512 / 64), 128, 0, stream>>>(xbf, wbuf, kf32, M,  512, 4096);
  cvtW(wv, (size_t)512 * 4096);
  gemm_bf16_wmma<<<dim3(M / 64,  512 / 64), 128, 0, stream>>>(xbf, wbuf, vf32, M,  512, 4096);

  // 3) RoPE + head-major relayout (bf16)
  {
    int tot = BB * SS * (NHH + NKVV) * 32;
    rope_kernel<<<(tot + 255) / 256, 256, 0, stream>>>(qf32, kf32, qbf, kfbf);
  }
  { int tot = BB * SS * NKVV * DD; vtrans_kernel<<<(tot + 255) / 256, 256, 0, stream>>>(vf32, vfbf); }
  { int n4 = BB * NHH * SS * DD / 4;
    cvt_bf16x4_kernel<<<(n4 + 255) / 256, 256, 0, stream>>>((const float4*)cache_k, (uint2*)k0bf, n4); }
  { int tot = BB * NHH * SS * DD;  v0t_kernel<<<(tot + 255) / 256, 256, 0, stream>>>(cache_v, v0t); }

  // 4) flash attention (WMMA) with diagonal extras
  flash_attn_kernel<<<dim3(SS / 16, NHH, BB), 32, 0, stream>>>(
      qbf, k0bf, v0t, cache_k, cache_v, kfbf, vfbf, attnb);

  // 5) output projection + residual
  cvtW(wo, (size_t)2048 * 2048);
  gemm_bf16_wmma<<<dim3(M / 64, 2048 / 64), 128, 0, stream>>>(attnb, wbuf, t0, M, 2048, 2048);
  { int tot = M * 2048; add_kernel<<<(tot + 255) / 256, 256, 0, stream>>>(hidden, t0, h2, tot); }

  // 6) MLP (WMMA): gate, up, silu*mul, down
  rmsnorm_kernel<<<M, 256, 0, stream>>>(h2, w_post_ln, hnbf);
  cvtW(wg, (size_t)II * HH);
  gemm_bf16_wmma<<<dim3(M / 64, 8192 / 64), 128, 0, stream>>>(hnbf, wbuf, t0, M, 8192, 2048);
  cvtW(wu, (size_t)II * HH);
  gemm_bf16_wmma<<<dim3(M / 64, 8192 / 64), 128, 0, stream>>>(hnbf, wbuf, t1, M, 8192, 2048);
  { int tot = M * 8192; silu_mul_kernel<<<(tot + 255) / 256, 256, 0, stream>>>(t0, t1, act, tot); }
  cvtW(wd, (size_t)HH * II);
  gemm_bf16_wmma<<<dim3(M / 64, 2048 / 64), 128, 0, stream>>>(act, wbuf, t0, M, 2048, 8192);

  // 7) final residual -> d_out (f32)
  { int tot = M * 2048; add_kernel<<<(tot + 255) / 256, 256, 0, stream>>>(h2, t0, out, tot); }
}
